// MHA_decoder_17403207483620
// MI455X (gfx1250) — matrix operand order repaired and checked by reading, not compiled, not run on hardware
//
#include <hip/hip_runtime.h>
#include <hip/hip_bf16.h>

// ---------------------------------------------------------------------------
// MHA decoder for MI455X (gfx1250, wave32, WMMA).
// All GEMMs + both attention einsums run on v_wmma_f32_16x16x32_bf16.
// ---------------------------------------------------------------------------

typedef __bf16 bf16;
typedef __attribute__((ext_vector_type(16))) __bf16 v16bf;
typedef __attribute__((ext_vector_type(8)))  __bf16 v8bf;
typedef __attribute__((ext_vector_type(8)))  float  v8f;

#define DEV_INLINE __device__ __forceinline__

DEV_INLINE v8f wmma_bf16(v16bf a, v16bf b, v8f c) {
    // (neg_a, A, neg_b, B, c_mod, C, reuse_a, reuse_b)
    return __builtin_amdgcn_wmma_f32_16x16x32_bf16(false, a, false, b, (short)0, c, false, false);
}

// A-matrix fragment (16x32 bf16). Lane l (half = l>>4, r = l&15) holds row M=r.
// half==0: v[0..15]=K0..15
// half==1: v[0..15]=K16..31
DEV_INLINE v16bf load_afrag(const bf16* __restrict__ rowPtr, int half) {
    v8bf lo = *(const v8bf*)(rowPtr + half * 16);
    v8bf hi = *(const v8bf*)(rowPtr + half * 16 + 8);
    return __builtin_shufflevector(lo, hi, 0,1,2,3,4,5,6,7,8,9,10,11,12,13,14,15);
}

// ---------------------------------------------------------------------------
// fp32 -> bf16 conversion
// ---------------------------------------------------------------------------
__global__ void cvt_f32_bf16(const float* __restrict__ in, bf16* __restrict__ out, int n) {
    int i = blockIdx.x * blockDim.x + threadIdx.x;
    if (i < n) out[i] = (bf16)in[i];
}

// ---------------------------------------------------------------------------
// GEMM: out = A @ W^T + bias.  A: (M,K) bf16 row-major, W: (N,K) bf16 row-major.
// Each wave computes a 32(M) x 64(N) tile: all 6 fragments loaded up front
// (one clause / one wait), then 8 WMMAs per K=32 step.
// MODE 0: bf16 out (M,N) row-major
// MODE 1: bf16 out transposed (N,M)   (V: makes attention B-frags contiguous)
// MODE 2: fp32 out (M,N) row-major    (final projection into d_out)
// ---------------------------------------------------------------------------
template<int MODE>
__global__ __launch_bounds__(128) void gemm_wmma(
    const bf16* __restrict__ A, const bf16* __restrict__ W,
    const float* __restrict__ bias, void* __restrict__ out,
    int M, int K, int N)
{
    const int lane = threadIdx.x & 31;
    const int wave = threadIdx.x >> 5;
    const int half = lane >> 4;
    const int r    = lane & 15;

    const int mTile = (blockIdx.y * 4 + wave) * 32;
    const int nTile = blockIdx.x * 64;
    if (mTile >= M) return; // wave-uniform

    v8f acc[2][4];
    #pragma unroll
    for (int mi = 0; mi < 2; ++mi)
        #pragma unroll
        for (int nj = 0; nj < 4; ++nj) acc[mi][nj] = (v8f){};

    const bf16* aRow0 = A + (size_t)(mTile + r) * K;
    const bf16* aRow1 = aRow0 + (size_t)16 * K;
    const bf16* wRow  = W + (size_t)(nTile + r) * K + half * 16;

    for (int kb = 0; kb < K; kb += 32) {
        // ---- batch all fragment loads, then compute ----
        v16bf a0 = load_afrag(aRow0 + kb, half);
        v16bf a1 = load_afrag(aRow1 + kb, half);
        v16bf b[4];
        #pragma unroll
        for (int nj = 0; nj < 4; ++nj)
            b[nj] = *(const v16bf*)(wRow + (size_t)(16 * nj) * K + kb);
        #pragma unroll
        for (int nj = 0; nj < 4; ++nj)
            acc[0][nj] = wmma_bf16(a0, b[nj], acc[0][nj]);
        #pragma unroll
        for (int nj = 0; nj < 4; ++nj)
            acc[1][nj] = wmma_bf16(a1, b[nj], acc[1][nj]);
    }

    // C layout: VGPR i -> row (tile + half*8 + i), col = nSub + lane.r
    #pragma unroll
    for (int mi = 0; mi < 2; ++mi) {
        #pragma unroll
        for (int i = 0; i < 8; ++i) {
            const int row = mTile + 16 * mi + half * 8 + i;
            #pragma unroll
            for (int nj = 0; nj < 4; ++nj) {
                const int col = nTile + 16 * nj + r;
                const float v = acc[mi][nj][i] + bias[col];
                if constexpr (MODE == 0)      ((bf16*)out)[(size_t)row * N + col] = (bf16)v;
                else if constexpr (MODE == 1) ((bf16*)out)[(size_t)col * M + row] = (bf16)v;
                else                          ((float*)out)[(size_t)row * N + col] = v;
            }
        }
    }
}

// ---------------------------------------------------------------------------
// RoPE (partial rotary, ROT=32 of HS=64): applied in place to bf16 q and k.
// rope: (2, T, 32) fp32: cos then sin.
// ---------------------------------------------------------------------------
__global__ void rope_apply(bf16* __restrict__ q, bf16* __restrict__ k,
                           const float* __restrict__ rope,
                           int BT, int T, int D, int H)
{
    int idx = blockIdx.x * blockDim.x + threadIdx.x;
    if (idx >= BT * H * 16) return;
    const int d   = idx & 15;
    const int h   = (idx >> 4) % H;
    const int row = idx / (16 * H);
    const int t   = row % T;
    const int T32 = T * 32;
    const float c1 = rope[t * 32 + d];
    const float s1 = rope[T32 + t * 32 + d];
    const float c2 = rope[t * 32 + 16 + d];
    const float s2 = rope[T32 + t * 32 + 16 + d];
    const size_t base = (size_t)row * D + h * 64;

    float q1 = (float)q[base + d], q2 = (float)q[base + 16 + d];
    q[base + d]      = (bf16)(q1 * c1 - q2 * s1);
    q[base + 16 + d] = (bf16)(q2 * c2 + q1 * s2);

    float k1 = (float)k[base + d], k2 = (float)k[base + 16 + d];
    k[base + d]      = (bf16)(k1 * c1 - k2 * s1);
    k[base + 16 + d] = (bf16)(k2 * c2 + k1 * s2);
}

// ---------------------------------------------------------------------------
// Flash attention, one wave per (b, h, 16-row q tile); 64 keys per iteration.
// Q:  (B*Tq, D) bf16     K: (B*Lk, D) bf16     Vt: (D, B*Lk) bf16 (transposed)
// Y:  (B*Tq, D) bf16.    HS = 64; 1/sqrt(HS) pre-folded into the Q fragment.
// All 8 K-fragments are loaded before the 8 QK WMMAs (one clause, one wait),
// WMMAs issued as independent chains; same for P@V. Next tiles prefetched.
// ---------------------------------------------------------------------------
__global__ __launch_bounds__(128) void flash_attn(
    const bf16* __restrict__ Q, const bf16* __restrict__ Km,
    const bf16* __restrict__ Vt, bf16* __restrict__ Y,
    int Tq, int Lk, int D, int H, int NV)
{
    const int lane = threadIdx.x & 31;
    const int wave = threadIdx.x >> 5;
    const int half = lane >> 4;
    const int r    = lane & 15;

    const int widx      = blockIdx.x * 4 + wave;
    const int qtPerHead = Tq >> 4;
    const int qt = widx % qtPerHead;
    const int h  = (widx / qtPerHead) % H;
    const int b  = widx / (qtPerHead * H);
    const int hbase = h * 64;
    const size_t qRow0 = (size_t)b * Tq + qt * 16;
    const size_t kRow0 = (size_t)b * Lk;

    __shared__ bf16 plds[4][16 * 64];   // per-wave P staging (C-layout -> A-layout)
    bf16* P = plds[wave];

    // Q fragments (A: 16x32 per HS chunk), with softmax scale folded in.
    const bf16* qRow = Q + (qRow0 + r) * D + hbase;
    v16bf aq0 = load_afrag(qRow,      half);
    v16bf aq1 = load_afrag(qRow + 32, half);
    #pragma unroll
    for (int e = 0; e < 16; ++e) {
        aq0[e] = (bf16)((float)aq0[e] * 0.125f);
        aq1[e] = (bf16)((float)aq1[e] * 0.125f);
    }

    // Hoisted pointers (strength-reduced addressing in the key loop).
    const bf16* kPtr  = Km + (kRow0 + r) * D + hbase + half * 16;
    const bf16* vPtr0 = Vt + (size_t)(hbase + r) * NV + kRow0 + half * 16;
    const size_t vStride16 = (size_t)16 * NV;

    v8f o[4];
    #pragma unroll
    for (int j = 0; j < 4; ++j) o[j] = (v8f){};
    float mstat[8], lstat[8];
    #pragma unroll
    for (int i = 0; i < 8; ++i) { mstat[i] = -1e30f; lstat[i] = 0.0f; }

    for (int s = 0; s < Lk; s += 64) {
        // ---- prefetch next 64-key K/V tiles into cache ----
        if (s + 64 < Lk) {
            __builtin_prefetch(kPtr + (size_t)(s + 64) * D, 0, 1);
            __builtin_prefetch(vPtr0 + s + 64, 0, 1);
        }

        // ---- load ALL K fragments (one clause), then 8 QK WMMAs ----
        v16bf bkLo[4], bkHi[4];
        #pragma unroll
        for (int t = 0; t < 4; ++t) {
            const bf16* kR = kPtr + (size_t)(s + 16 * t) * D;
            bkLo[t] = *(const v16bf*)(kR);
            bkHi[t] = *(const v16bf*)(kR + 32);
        }
        v8f st[4];
        #pragma unroll
        for (int t = 0; t < 4; ++t) st[t] = wmma_bf16(aq0, bkLo[t], (v8f){});
        #pragma unroll
        for (int t = 0; t < 4; ++t) st[t] = wmma_bf16(aq1, bkHi[t], st[t]);

        // ---- row max, stage-major xor reduction within each 16-lane half ----
        float mx[8];
        #pragma unroll
        for (int i = 0; i < 8; ++i)
            mx[i] = fmaxf(fmaxf(st[0][i], st[1][i]), fmaxf(st[2][i], st[3][i]));
        #pragma unroll
        for (int msk = 8; msk >= 1; msk >>= 1) {
            float t[8];
            #pragma unroll
            for (int i = 0; i < 8; ++i) t[i] = __shfl_xor(mx[i], msk, 16);
            #pragma unroll
            for (int i = 0; i < 8; ++i) mx[i] = fmaxf(mx[i], t[i]);
        }

        // ---- exp + P staging + partial row sums ----
        float alpha[8], rs[8];
        #pragma unroll
        for (int i = 0; i < 8; ++i) {
            const float mnew = fmaxf(mstat[i], mx[i]);
            alpha[i] = __expf(mstat[i] - mnew);
            mstat[i] = mnew;
            const float p0 = __expf(st[0][i] - mnew);
            const float p1 = __expf(st[1][i] - mnew);
            const float p2 = __expf(st[2][i] - mnew);
            const float p3 = __expf(st[3][i] - mnew);
            rs[i] = (p0 + p1) + (p2 + p3);
            bf16* pr = P + (half * 8 + i) * 64 + r;
            pr[0]  = (bf16)p0;
            pr[16] = (bf16)p1;
            pr[32] = (bf16)p2;
            pr[48] = (bf16)p3;
        }
        #pragma unroll
        for (int msk = 8; msk >= 1; msk >>= 1) {
            float t[8];
            #pragma unroll
            for (int i = 0; i < 8; ++i) t[i] = __shfl_xor(rs[i], msk, 16);
            #pragma unroll
            for (int i = 0; i < 8; ++i) rs[i] += t[i];
        }
        #pragma unroll
        for (int i = 0; i < 8; ++i) lstat[i] = lstat[i] * alpha[i] + rs[i];
        #pragma unroll
        for (int j = 0; j < 4; ++j)
            #pragma unroll
            for (int i = 0; i < 8; ++i) o[j][i] *= alpha[i];

        // ---- V fragments + P (via LDS) loaded up front, then 8 PV WMMAs ----
        v16bf bvLo[4], bvHi[4];
        #pragma unroll
        for (int j = 0; j < 4; ++j) {
            const bf16* vc = vPtr0 + (size_t)j * vStride16 + s;
            bvLo[j] = *(const v16bf*)(vc);
            bvHi[j] = *(const v16bf*)(vc + 32);
        }
        const v16bf ap0 = load_afrag(P + r * 64,      half);
        const v16bf ap1 = load_afrag(P + r * 64 + 32, half);
        #pragma unroll
        for (int j = 0; j < 4; ++j) o[j] = wmma_bf16(ap0, bvLo[j], o[j]);
        #pragma unroll
        for (int j = 0; j < 4; ++j) o[j] = wmma_bf16(ap1, bvHi[j], o[j]);
    }

    // ---- normalize + store ----
    #pragma unroll
    for (int i = 0; i < 8; ++i) {
        const float inv = 1.0f / lstat[i];
        bf16* yr = Y + (qRow0 + half * 8 + i) * D + hbase + r;
        yr[0]  = (bf16)(o[0][i] * inv);
        yr[16] = (bf16)(o[1][i] * inv);
        yr[32] = (bf16)(o[2][i] * inv);
        yr[48] = (bf16)(o[3][i] * inv);
    }
}

// ---------------------------------------------------------------------------
// Host orchestration
// ---------------------------------------------------------------------------
extern "C" void kernel_launch(void* const* d_in, const int* in_sizes, int n_in,
                              void* d_out, int out_size, void* d_ws, size_t ws_size,
                              hipStream_t stream) {
    (void)in_sizes; (void)n_in; (void)out_size; (void)ws_size;
    const float* x      = (const float*)d_in[0];
    const float* memory = (const float*)d_in[1];
    const float* rope   = (const float*)d_in[2];
    const float* Wq     = (const float*)d_in[3];
    const float* bq     = (const float*)d_in[4];
    const float* Wk     = (const float*)d_in[5];
    const float* bk     = (const float*)d_in[6];
    const float* Wv     = (const float*)d_in[7];
    const float* bv     = (const float*)d_in[8];
    const float* Wo     = (const float*)d_in[9];
    const float* bo     = (const float*)d_in[10];

    constexpr int B = 4, T = 1024, L = 1024, D = 1024, H = 16;
    const int BT = B * T, BL = B * L;

    char* ws = (char*)d_ws;
    size_t off = 0;
    auto carve = [&](size_t bytes) -> char* {
        char* p = ws + off;
        off += (bytes + 255) & ~(size_t)255;
        return p;
    };
    bf16* xb  = (bf16*)carve((size_t)BT * D * 2);
    bf16* mb  = (bf16*)carve((size_t)BL * D * 2);
    bf16* wqb = (bf16*)carve((size_t)D * D * 2);
    bf16* wkb = (bf16*)carve((size_t)D * D * 2);
    bf16* wvb = (bf16*)carve((size_t)D * D * 2);
    bf16* wob = (bf16*)carve((size_t)D * D * 2);
    bf16* qb  = (bf16*)carve((size_t)BT * D * 2);
    bf16* kb_ = (bf16*)carve((size_t)BL * D * 2);
    bf16* vtb = (bf16*)carve((size_t)BL * D * 2);
    bf16* y1  = (bf16*)carve((size_t)BT * D * 2);
    bf16* y2  = xb; // x no longer needed after stage-1 projections

    const int cvtThr = 256;
    cvt_f32_bf16<<<(BT * D + cvtThr - 1) / cvtThr, cvtThr, 0, stream>>>(x, xb, BT * D);
    cvt_f32_bf16<<<(BL * D + cvtThr - 1) / cvtThr, cvtThr, 0, stream>>>(memory, mb, BL * D);
    cvt_f32_bf16<<<(D * D + cvtThr - 1) / cvtThr, cvtThr, 0, stream>>>(Wq, wqb, D * D);
    cvt_f32_bf16<<<(D * D + cvtThr - 1) / cvtThr, cvtThr, 0, stream>>>(Wk, wkb, D * D);
    cvt_f32_bf16<<<(D * D + cvtThr - 1) / cvtThr, cvtThr, 0, stream>>>(Wv, wvb, D * D);
    cvt_f32_bf16<<<(D * D + cvtThr - 1) / cvtThr, cvtThr, 0, stream>>>(Wo, wob, D * D);

    // 128 threads = 4 waves; each wave owns a 32x64 tile -> block tile 128x64.
    const dim3 blk(128);
    const dim3 gProj(D / 64, BT / 128);
    const dim3 gProjL(D / 64, BL / 128);

    // ---- stage 1: self attention on x ----
    gemm_wmma<0><<<gProj, blk, 0, stream>>>(xb, wqb, bq, qb,  BT, D, D);
    gemm_wmma<0><<<gProj, blk, 0, stream>>>(xb, wkb, bk, kb_, BT, D, D);
    gemm_wmma<1><<<gProj, blk, 0, stream>>>(xb, wvb, bv, vtb, BT, D, D);

    rope_apply<<<(BT * H * 16 + 255) / 256, 256, 0, stream>>>(qb, kb_, rope, BT, T, D, H);

    const int nWaves = B * H * (T / 16);
    flash_attn<<<nWaves / 4, blk, 0, stream>>>(qb, kb_, vtb, y1, T, T, D, H, BT);

    // ---- stage 2: cross attention with memory ----
    gemm_wmma<0><<<gProj,  blk, 0, stream>>>(y1, wqb, bq, qb,  BT, D, D);
    gemm_wmma<0><<<gProjL, blk, 0, stream>>>(mb, wkb, bk, kb_, BL, D, D);
    gemm_wmma<1><<<gProjL, blk, 0, stream>>>(mb, wvb, bv, vtb, BL, D, D);

    flash_attn<<<nWaves / 4, blk, 0, stream>>>(qb, kb_, vtb, y2, T, L, D, H, BL);

    // ---- output projection (fp32 into d_out) ----
    gemm_wmma<2><<<gProj, blk, 0, stream>>>(y2, wob, bo, (float*)d_out, BT, D, D);
}
